// MetaS4History_70506183131280
// MI455X (gfx1250) — matrix operand order repaired and checked by reading, not compiled, not run on hardware
//
#include <hip/hip_runtime.h>
#include <hip/hip_bf16.h>
#include <math.h>

// ---------------- problem constants ----------------
constexpr int Bn  = 16;
constexpr int Ln  = 2048;
constexpr int Hn  = 512;
constexpr int Nn  = 64;
constexpr int Tc  = 32;            // chunk length (== WMMA K for bf16)
constexpr int NCc = Ln / Tc;       // 64 chunks per sequence
constexpr int BLn = Bn * Ln;       // 32768 columns

// ---------------- vector types ----------------
typedef __bf16        v16bf __attribute__((ext_vector_type(16)));
typedef float         v8f   __attribute__((ext_vector_type(8)));
typedef unsigned int  v8u   __attribute__((ext_vector_type(8)));
typedef unsigned int  v4u   __attribute__((ext_vector_type(4)));
typedef int           v8i   __attribute__((ext_vector_type(8)));
typedef int           v4i   __attribute__((ext_vector_type(4)));

// ---------------- bf16 helpers (RNE) ----------------
__device__ __forceinline__ unsigned short f2bf(float f) {
  unsigned int u = __builtin_bit_cast(unsigned int, f);
  unsigned int r = u + 0x7FFFu + ((u >> 16) & 1u);
  return (unsigned short)(r >> 16);
}
__device__ __forceinline__ float bf2f(unsigned short s) {
  unsigned int u = ((unsigned int)s) << 16;
  return __builtin_bit_cast(float, u);
}

// ---------------- WMMA wrapper ----------------
__device__ __forceinline__ v8f wmma_bf16(v8u a, v8u b, v8f c) {
  return __builtin_amdgcn_wmma_f32_16x16x32_bf16(
      false, __builtin_bit_cast(v16bf, a),
      false, __builtin_bit_cast(v16bf, b),
      (short)0, c, false, false);
}

// A-operand (16x32 bf16) lane/VGPR -> (m,k) mapping per CDNA5 ISA 7.12.2:
//   lanes 0-15: V0..V3 hold K=0..7 pairs, V4..V7 hold K=16..23
//   lanes16-31: V0..V3 hold K=8..15,      V4..V7 hold K=24..31 ; m = lane&15
__device__ __forceinline__ int a16_k0(int lane, int v) {
  return ((v & 3) * 2) + ((lane >> 4) * 8) + ((v >> 2) * 16);
}
// B-operand (32x16 bf16): same K order as A (lanes0-15 K=0..7,16..23; lanes16-31 K=8..15,24..31); n = lane&15
__device__ __forceinline__ int b16_k0(int lane, int v) {
  return ((v & 3) * 2) + ((lane >> 4) * 8) + ((v >> 2) * 16);
}

// load a pre-packed A fragment: 8 consecutive dwords per lane
__device__ __forceinline__ v8u load_frag8(const unsigned int* p, int lane) {
  v8u r;
  const unsigned int* q = p + lane * 8;
#pragma unroll
  for (int i = 0; i < 8; ++i) r[i] = q[i];
  return r;
}

// B fragment where consecutive k are contiguous in memory: elem(k,n) at base[ (col)*stride + k ]
__device__ __forceinline__ v8u frag_b_pairs(const unsigned short* base, int stride,
                                            int col0, int kbase, int lane) {
  v8u r;
  int c = col0 + (lane & 15);
#pragma unroll
  for (int v = 0; v < 8; ++v) {
    int k0 = kbase + b16_k0(lane, v);
    r[v] = *(const unsigned int*)(base + c * stride + k0);
  }
  return r;
}

// B fragment from row-major [k][n] tile (k pairs strided): two u16 loads per dword
__device__ __forceinline__ v8u frag_b_rowmajor(const unsigned short* base, int ldn,
                                               int col0, int lane) {
  v8u r;
  int n = col0 + (lane & 15);
#pragma unroll
  for (int v = 0; v < 8; ++v) {
    int k0 = b16_k0(lane, v);
    unsigned int lo = base[(k0 + 0) * ldn + n];
    unsigned int hi = base[(k0 + 1) * ldn + n];
    r[v] = lo | (hi << 16);
  }
  return r;
}

// A fragment from row-major [m][k] bf16 (consecutive k contiguous)
__device__ __forceinline__ v8u frag_a_rowmajor(const unsigned short* base, int ldk,
                                               int row0, int kbase, int lane) {
  v8u r;
  int m = row0 + (lane & 15);
#pragma unroll
  for (int v = 0; v < 8; ++v) {
    int k0 = kbase + a16_k0(lane, v);
    r[v] = *(const unsigned int*)(base + (size_t)m * ldk + k0);
  }
  return r;
}

// ---------------- TDM: 2D bf16 tile -> LDS (CDNA5 Tensor Data Mover) ----------------
// Descriptor per ISA 8.3/8.4 (D# group0/group1). global_addr = tile start,
// tensor dims set huge so no OOB zeroing, stride in elements.
// This toolchain exposes the 6-arg builtin: (v4u, v8i, v4i, v4i, v8i, i32 cpol).
__device__ __forceinline__ void tdm_load_2d_bf16(const void* gsrc, unsigned lds_off,
                                                 unsigned tile_w, unsigned tile_h,
                                                 unsigned long long stride_elems) {
  unsigned long long ga = (unsigned long long)(size_t)gsrc;
  v4u g0;
  g0[0] = 1u;                                        // count=1, user descriptor
  g0[1] = lds_off;                                   // lds_addr (bytes)
  g0[2] = (unsigned)(ga & 0xFFFFFFFFull);            // global_addr[31:0]
  g0[3] = (unsigned)((ga >> 32) & 0x01FFFFFFull)     // global_addr[56:32]
          | 0x80000000u;                             // type=2 ("image")
  unsigned long long dim0 = 1ull << 30, dim1 = 1ull << 20;
  v8i g1;
  g1[0] = 0x00010000;                                // data_size=1 (2 bytes), no mask
  g1[1] = (int)((dim0 & 0xFFFFull) << 16);           // tensor_dim0[15:0]
  g1[2] = (int)(((dim0 >> 16) & 0xFFFFull) | ((dim1 & 0xFFFFull) << 16));
  g1[3] = (int)(((dim1 >> 16) & 0xFFFFull) | ((unsigned long long)tile_w << 16));
  g1[4] = (int)(tile_h & 0xFFFFu);                   // tile_dim1 (tile_dim2=0)
  g1[5] = (int)(stride_elems & 0xFFFFFFFFull);       // tensor_dim0_stride[31:0]
  g1[6] = (int)((stride_elems >> 32) & 0xFFFFull);
  g1[7] = 0;
  v4i g2 = {0, 0, 0, 0};
  v4i g3 = {0, 0, 0, 0};
  v8i g4 = {0, 0, 0, 0, 0, 0, 0, 0};                 // extra group (6-arg form)
  __builtin_amdgcn_tensor_load_to_lds(g0, g1, g2, g3, g4, 0);
}

// =====================================================================
// K1: per (h) mode setup. Computes discretized poles w = exp(dt*A),
// coeff = C (w-1)/A, then packs (in WMMA A-fragment layout, bf16):
//   Pw : N x T   with P[n][t]  = w^(T-1-t)         (chunk summary)
//   Ew : T x N   with E0=2Re(coeff w^(t+1)), E1=-2Im(...)  (state expansion)
//   Tw : T x T   lower-tri Toeplitz of K[d]=2Re(sum coeff w^d)
//   wT : w^T per (h,n), f32                         (chunk scan)
// =====================================================================
__global__ __launch_bounds__(64) void k_setup(
    const float* __restrict__ ldt, const float* __restrict__ lAre,
    const float* __restrict__ Aim, const float* __restrict__ Cre,
    const float* __restrict__ Cim,
    unsigned int* __restrict__ Pw, unsigned int* __restrict__ Ew,
    unsigned int* __restrict__ Tw, float* __restrict__ wT) {
  int h = blockIdx.x;
  int n = threadIdx.x;  // 64 threads, one per mode
  __shared__ float Pr[Tc][Nn], Pi[Tc][Nn], Er[Tc][Nn], Ei[Tc][Nn];
  __shared__ float Ktab[Tc], Cr0[Nn];

  float dt = __expf(ldt[h]);
  float ar = -__expf(lAre[h * Nn + n]);
  float ai = Aim[h * Nn + n];
  float cr = Cre[h * Nn + n], ci = Cim[h * Nn + n];
  float em = __expf(dt * ar);
  float wr = em * __cosf(dt * ai);
  float wi = em * __sinf(dt * ai);
  // coeff = C * (w - 1) / A
  float den = ar * ar + ai * ai;
  float tr = wr - 1.0f, ti = wi;
  float qr = (tr * ar + ti * ai) / den;
  float qi = (ti * ar - tr * ai) / den;
  float kr = cr * qr - ci * qi;
  float ki = cr * qi + ci * qr;
  Cr0[n] = kr;

  // ascending powers of w; P[t] = w^(T-1-t)
  float pr = 1.f, pi = 0.f;
#pragma unroll
  for (int d = 0; d < Tc; ++d) {
    Pr[Tc - 1 - d][n] = pr;
    Pi[Tc - 1 - d][n] = pi;
    float nr = pr * wr - pi * wi;
    pi = pr * wi + pi * wr;
    pr = nr;
  }
  // now (pr,pi) = w^T
  wT[(h * Nn + n) * 2 + 0] = pr;
  wT[(h * Nn + n) * 2 + 1] = pi;

  // E[t] = coeff * w^(t+1)
  float sr = kr * wr - ki * wi;
  float si = kr * wi + ki * wr;
#pragma unroll
  for (int t = 0; t < Tc; ++t) {
    Er[t][n] = sr;
    Ei[t][n] = si;
    float nr = sr * wr - si * wi;
    si = sr * wi + si * wr;
    sr = nr;
  }
  __syncthreads();

  // K[d] = 2 Re(sum_n coeff w^d); d=0 from coeff, d>=1 from Er[d-1]
  if (n < Tc) {
    float s = 0.f;
    if (n == 0) {
      for (int j = 0; j < Nn; ++j) s += Cr0[j];
    } else {
      for (int j = 0; j < Nn; ++j) s += Er[n - 1][j];
    }
    Ktab[n] = 2.0f * s;
  }
  __syncthreads();

  // ---- pack P fragments: [part(2)][rt(4)] x 256 dwords; A rows = modes, k = t
  unsigned int* PwH = Pw + (size_t)h * 2048;
  for (int idx = threadIdx.x; idx < 2048; idx += 64) {
    int part = idx >> 10;
    int rt = (idx >> 8) & 3;
    int e = idx & 255, lane = e >> 3, v = e & 7;
    int m = lane & 15;
    int k0 = a16_k0(lane, v);
    int nrow = rt * 16 + m;
    float f0 = part ? Pi[k0 + 0][nrow] : Pr[k0 + 0][nrow];
    float f1 = part ? Pi[k0 + 1][nrow] : Pr[k0 + 1][nrow];
    PwH[idx] = (unsigned)f2bf(f0) | ((unsigned)f2bf(f1) << 16);
  }
  // ---- pack E fragments: [part(2)][kt(2)][rt(2)]; A rows = t, k = mode
  unsigned int* EwH = Ew + (size_t)h * 2048;
  for (int idx = threadIdx.x; idx < 2048; idx += 64) {
    int part = idx >> 10;
    int kt = (idx >> 9) & 1;
    int rt = (idx >> 8) & 1;
    int e = idx & 255, lane = e >> 3, v = e & 7;
    int t = rt * 16 + (lane & 15);
    int k0 = kt * 32 + a16_k0(lane, v);
    float sgn = part ? -2.f : 2.f;
    float f0 = sgn * (part ? Ei[t][k0 + 0] : Er[t][k0 + 0]);
    float f1 = sgn * (part ? Ei[t][k0 + 1] : Er[t][k0 + 1]);
    EwH[idx] = (unsigned)f2bf(f0) | ((unsigned)f2bf(f1) << 16);
  }
  // ---- pack Toeplitz fragments: [rt(2)]; A[t][j] = (j<=t) ? K[t-j] : 0
  unsigned int* TwH = Tw + (size_t)h * 512;
  for (int idx = threadIdx.x; idx < 512; idx += 64) {
    int rt = idx >> 8;
    int e = idx & 255, lane = e >> 3, v = e & 7;
    int t = rt * 16 + (lane & 15);
    int j0 = a16_k0(lane, v);
    float f0 = (j0 + 0 <= t) ? Ktab[t - j0 - 0] : 0.f;
    float f1 = (j0 + 1 <= t) ? Ktab[t - j0 - 1] : 0.f;
    TwH[idx] = (unsigned)f2bf(f0) | ((unsigned)f2bf(f1) << 16);
  }
}

// =====================================================================
// K1b: weights f32 -> bf16 (row-major (o,h), h contiguous => direct A frags)
// =====================================================================
__global__ void k_packw(const float* __restrict__ W, unsigned short* __restrict__ Wbf) {
  int i = blockIdx.x * 256 + threadIdx.x;
  if (i < 2 * Hn * Hn) Wbf[i] = f2bf(W[i]);
}

// =====================================================================
// K2: fused S4 convolution for one (h,b): chunked scan, all-WMMA.
// 256 threads = 8 waves. LDS: u (4KB) + p (32KB) + s_bf16 (16KB) = 52KB.
// =====================================================================
__global__ __launch_bounds__(256) void k_conv(
    const unsigned short* __restrict__ Ubf, const unsigned int* __restrict__ Pw,
    const unsigned int* __restrict__ Ew, const unsigned int* __restrict__ Tw,
    const float* __restrict__ wT, const float* __restrict__ Dv,
    unsigned short* __restrict__ Ybf) {
  int h = blockIdx.x, b = blockIdx.y;
  int tid = threadIdx.x;
  int lane = tid & 31, wv = tid >> 5;

  __shared__ unsigned short uL[Ln];               // bf16 input, l-contiguous
  __shared__ float psr[NCc * Nn], psi[NCc * Nn];  // chunk summaries (f32)
  __shared__ unsigned short sre[NCc * Nn], sim_[NCc * Nn];  // boundary states bf16

  const unsigned short* src = Ubf + (size_t)h * BLn + (size_t)b * Ln;
  for (int i = tid; i < Ln / 2; i += 256)
    ((unsigned int*)uL)[i] = ((const unsigned int*)src)[i];
  __syncthreads();

  // ---- chunk summaries: p[n, c] = sum_t w^(T-1-t) u[cT+t]   (32 WMMA / wg)
  {
    int rt = wv & 3, part = wv >> 2;
    v8u a = load_frag8(Pw + (size_t)h * 2048 + (part * 4 + rt) * 256, lane);
    float* dst = part ? psi : psr;
#pragma unroll
    for (int ct = 0; ct < 4; ++ct) {
      v8u bu = frag_b_pairs(uL, Tc, ct * 16, 0, lane);
      v8f acc = {0.f, 0.f, 0.f, 0.f, 0.f, 0.f, 0.f, 0.f};
      acc = wmma_bf16(a, bu, acc);
      int c = ct * 16 + (lane & 15);
      int mhi = (lane >> 4) * 8;
#pragma unroll
      for (int r = 0; r < 8; ++r) dst[c * Nn + (rt * 16 + mhi + r)] = acc[r];
    }
  }
  __syncthreads();

  // ---- chunk-boundary scan (64 modes, 64 sequential chunk steps, f32)
  if (tid < Nn) {
    int n = tid;
    float wtr = wT[(h * Nn + n) * 2 + 0];
    float wti = wT[(h * Nn + n) * 2 + 1];
    float sr = 0.f, si = 0.f;
    for (int c = 0; c < NCc; ++c) {
      sre[c * Nn + n] = f2bf(sr);
      sim_[c * Nn + n] = f2bf(si);
      float pr = psr[c * Nn + n], pi = psi[c * Nn + n];
      float nr = wtr * sr - wti * si + pr;
      si = wtr * si + wti * sr + pi;
      sr = nr;
    }
  }
  __syncthreads();

  // ---- y = Toep(K) @ u + E @ s_in, then gelu(y + D*u)   (40 WMMA / wg)
  {
    int rt = wv & 1, ct = wv >> 1;
    v8f acc = {0.f, 0.f, 0.f, 0.f, 0.f, 0.f, 0.f, 0.f};
    v8u at = load_frag8(Tw + (size_t)h * 512 + rt * 256, lane);
    v8u bu = frag_b_pairs(uL, Tc, ct * 16, 0, lane);
    acc = wmma_bf16(at, bu, acc);
    const unsigned int* EwH = Ew + (size_t)h * 2048;
#pragma unroll
    for (int part = 0; part < 2; ++part) {
      const unsigned short* sb = part ? sim_ : sre;
#pragma unroll
      for (int kt = 0; kt < 2; ++kt) {
        v8u ae = load_frag8(EwH + ((part * 2 + kt) * 2 + rt) * 256, lane);
        v8u bs = frag_b_pairs(sb, Nn, ct * 16, kt * 32, lane);
        acc = wmma_bf16(ae, bs, acc);
      }
    }
    float Dh = Dv[h];
    unsigned short* dst = Ybf + (size_t)h * BLn + (size_t)b * Ln;
    int c = ct * 16 + (lane & 15);
    int mhi = (lane >> 4) * 8;
#pragma unroll
    for (int r = 0; r < 8; ++r) {
      int t = rt * 16 + mhi + r;
      int l = c * Tc + t;
      float uval = bf2f(uL[l]);
      float v = acc[r] + Dh * uval;
      // tanh-approximate gelu (jax.nn.gelu default)
      float g = 0.5f * v * (1.0f + tanhf(0.7978845608f * (v + 0.044715f * v * v * v)));
      dst[l] = f2bf(g);
    }
  }
}

// =====================================================================
// K3: GLU output linear. z = W(1024x512) @ y + b ; out = z_top * sigmoid(z_bot)
// Block: 32 paired rows x 64 cols, 8 waves, K-loop with TDM-staged B tiles.
// =====================================================================
__global__ __launch_bounds__(256) void k_gemm(
    const unsigned short* __restrict__ Wbf, const float* __restrict__ bias,
    const unsigned short* __restrict__ Ybf, const unsigned short* __restrict__ resbf,
    unsigned short* __restrict__ outbf, float* __restrict__ outf) {
  int tid = threadIdx.x;
  int lane = tid & 31, wv = tid >> 5;
  int r0 = blockIdx.x * 32;
  int col0 = blockIdx.y * 64;

  __shared__ unsigned short Bt[32 * 64];  // staged Y tile [k][col], 4KB
  __shared__ float Zb[32 * 64];           // bottom-half z for GLU pairing, 8KB

  int rt = wv & 3;
  int cb = (wv >> 2) * 2;
  int rowbase = (rt < 2) ? (r0 + rt * 16) : (512 + r0 + (rt - 2) * 16);
  v8f acc0 = {0.f, 0.f, 0.f, 0.f, 0.f, 0.f, 0.f, 0.f};
  v8f acc1 = {0.f, 0.f, 0.f, 0.f, 0.f, 0.f, 0.f, 0.f};
  unsigned lds_off = (unsigned)(size_t)(void*)Bt;  // LDS aperture: low 32 bits = offset

  for (int k0 = 0; k0 < Hn; k0 += 32) {
    if (tid < 32) {  // wave 0 drives the Tensor Data Mover
      tdm_load_2d_bf16(Ybf + (size_t)k0 * BLn + col0, lds_off, 64, 32,
                       (unsigned long long)BLn);
      __builtin_amdgcn_s_wait_tensorcnt(0);
    }
    __syncthreads();
    v8u a = frag_a_rowmajor(Wbf, Hn, rowbase, k0, lane);
    v8u b0 = frag_b_rowmajor(Bt, 64, (cb + 0) * 16, lane);
    v8u b1 = frag_b_rowmajor(Bt, 64, (cb + 1) * 16, lane);
    acc0 = wmma_bf16(a, b0, acc0);
    acc1 = wmma_bf16(a, b1, acc1);
    __syncthreads();
  }

  // bottom-half waves stash z into LDS for GLU pairing
  if (rt >= 2) {
    int mb = (rt - 2) * 16 + (lane >> 4) * 8;
    int cl = cb * 16 + (lane & 15);
#pragma unroll
    for (int r = 0; r < 8; ++r) Zb[(mb + r) * 64 + cl] = acc0[r];
    cl += 16;
#pragma unroll
    for (int r = 0; r < 8; ++r) Zb[(mb + r) * 64 + cl] = acc1[r];
  }
  __syncthreads();

  if (rt < 2) {
    int hi = (lane >> 4) * 8;
#pragma unroll
    for (int half = 0; half < 2; ++half) {
      v8f acc = half ? acc1 : acc0;
      int cl = (cb + half) * 16 + (lane & 15);
#pragma unroll
      for (int r = 0; r < 8; ++r) {
        int mloc = rt * 16 + hi + r;  // 0..31 within block row-tile
        float zt = acc[r] + bias[r0 + mloc];
        float zb = Zb[mloc * 64 + cl] + bias[512 + r0 + mloc];
        float val = zt * (1.0f / (1.0f + __expf(-zb)));
        size_t gidx = (size_t)(r0 + mloc) * BLn + (size_t)(col0 + cl);
        if (resbf) val += bf2f(resbf[gidx]);
        if (outbf) outbf[gidx] = f2bf(val);
        if (outf) outf[gidx] = val;
      }
    }
  }
}

// =====================================================================
// Transpose kernels: (B,L,H) <-> (H, B*L)
// =====================================================================
__global__ __launch_bounds__(256) void k_trans_in(const float* __restrict__ x,
                                                  unsigned short* __restrict__ Ubf) {
  __shared__ float tile[32][33];
  int cbase = blockIdx.x * 32;  // BL dim
  int hbase = blockIdx.y * 32;  // H dim
  int tx = threadIdx.x, ty = threadIdx.y;
  for (int i = ty; i < 32; i += 8)
    tile[i][tx] = x[(size_t)(cbase + i) * Hn + hbase + tx];
  __syncthreads();
  for (int i = ty; i < 32; i += 8)
    Ubf[(size_t)(hbase + i) * BLn + cbase + tx] = f2bf(tile[tx][i]);
}

__global__ __launch_bounds__(256) void k_trans_out(const float* __restrict__ Uf,
                                                   float* __restrict__ out) {
  __shared__ float tile[32][33];
  int cbase = blockIdx.x * 32;
  int hbase = blockIdx.y * 32;
  int tx = threadIdx.x, ty = threadIdx.y;
  for (int i = ty; i < 32; i += 8)
    tile[i][tx] = Uf[(size_t)(hbase + i) * BLn + cbase + tx];
  __syncthreads();
  for (int i = ty; i < 32; i += 8)
    out[(size_t)(cbase + i) * Hn + hbase + tx] = tile[tx][i];
}

// =====================================================================
// workspace layout (bytes)
// =====================================================================
constexpr size_t SZ_ACT_BF = (size_t)Hn * BLn * 2;   // 32MB bf16 activations
constexpr size_t SZ_ACT_F  = (size_t)Hn * BLn * 4;   // 64MB f32
constexpr size_t SZ_WBF = (size_t)2 * Hn * Hn * 2;   // 1MB
constexpr size_t SZ_PW = (size_t)Hn * 2048 * 4;      // 4MB
constexpr size_t SZ_EW = (size_t)Hn * 2048 * 4;      // 4MB
constexpr size_t SZ_TW = (size_t)Hn * 512 * 4;       // 1MB
constexpr size_t SZ_WT = (size_t)Hn * Nn * 2 * 4;    // 256KB
constexpr size_t LPAR = SZ_WBF + SZ_PW + SZ_EW + SZ_TW + SZ_WT;

extern "C" void kernel_launch(void* const* d_in, const int* in_sizes, int n_in,
                              void* d_out, int out_size, void* d_ws, size_t ws_size,
                              hipStream_t stream) {
  (void)in_sizes; (void)n_in; (void)out_size; (void)ws_size;
  char* ws = (char*)d_ws;
  unsigned short* Ubf  = (unsigned short*)(ws);
  unsigned short* Ybf  = (unsigned short*)(ws + SZ_ACT_BF);
  unsigned short* U2bf = (unsigned short*)(ws + 2 * SZ_ACT_BF);
  float*          Uf   = (float*)(ws + 3 * SZ_ACT_BF);
  char* par = ws + 3 * SZ_ACT_BF + SZ_ACT_F;

  const float* x = (const float*)d_in[0];

  // input transpose (B,L,H) -> bf16 (H, B*L)
  k_trans_in<<<dim3(BLn / 32, Hn / 32), dim3(32, 8), 0, stream>>>(x, Ubf);

  for (int layer = 0; layer < 2; ++layer) {
    int base = 1 + 8 * layer;
    const float* ldt  = (const float*)d_in[base + 0];
    const float* lAre = (const float*)d_in[base + 1];
    const float* Aim  = (const float*)d_in[base + 2];
    const float* Cre  = (const float*)d_in[base + 3];
    const float* Cim  = (const float*)d_in[base + 4];
    const float* Dv   = (const float*)d_in[base + 5];
    const float* W    = (const float*)d_in[base + 6];
    const float* bv   = (const float*)d_in[base + 7];

    char* lp = par + (size_t)layer * LPAR;
    unsigned short* Wbf = (unsigned short*)(lp);
    unsigned int* Pw = (unsigned int*)(lp + SZ_WBF);
    unsigned int* Ew = (unsigned int*)(lp + SZ_WBF + SZ_PW);
    unsigned int* Tw = (unsigned int*)(lp + SZ_WBF + SZ_PW + SZ_EW);
    float* wT = (float*)(lp + SZ_WBF + SZ_PW + SZ_EW + SZ_TW);

    k_setup<<<Hn, 64, 0, stream>>>(ldt, lAre, Aim, Cre, Cim, Pw, Ew, Tw, wT);
    k_packw<<<(2 * Hn * Hn + 255) / 256, 256, 0, stream>>>(W, Wbf);

    const unsigned short* inbf = (layer == 0) ? Ubf : U2bf;
    k_conv<<<dim3(Hn, Bn), 256, 0, stream>>>(inbf, Pw, Ew, Tw, wT, Dv, Ybf);

    if (layer == 0) {
      // residual: out = glu + u  -> next layer's bf16 input
      k_gemm<<<dim3(Hn / 32, BLn / 64), 256, 0, stream>>>(
          Wbf, bv, Ybf, Ubf, U2bf, (float*)nullptr);
    } else {
      // final layer: f32 output, no residual
      k_gemm<<<dim3(Hn / 32, BLn / 64), 256, 0, stream>>>(
          Wbf, bv, Ybf, (const unsigned short*)nullptr,
          (unsigned short*)nullptr, Uf);
    }
  }

  // output transpose (H, B*L) -> (B,L,H)
  k_trans_out<<<dim3(BLn / 32, Hn / 32), dim3(32, 8), 0, stream>>>(Uf, (float*)d_out);
}
